// ProteinGCNNet_9036611191474
// MI455X (gfx1250) — matrix operand order repaired and checked by reading, not compiled, not run on hardware
//
#include <hip/hip_runtime.h>

// ---------------------------------------------------------------------------
// GCN (3x GCNConv + mean-pool + 2-layer MLP) for MI455X (gfx1250, wave32).
// GEMMs run on v_wmma_f32_16x16x32_f16 (f16 inputs, f32 accumulate).
// Edge aggregation = thread-per-(edge,feature) global_atomic_add_f32
// (memory-bound part: ~4.8GB irregular traffic, L2-resident at 192MB).
// ---------------------------------------------------------------------------

typedef __attribute__((ext_vector_type(16))) _Float16 v16h;
typedef __attribute__((ext_vector_type(8)))  _Float16 v8h;
typedef __attribute__((ext_vector_type(8)))  float    v8f;

#define NNODES 100000
#define NGRAPH 256

// ---------------- elementwise helpers ----------------

__global__ void k_fill(float* __restrict__ p, float v, long n) {
  for (long i = blockIdx.x * (long)blockDim.x + threadIdx.x; i < n;
       i += (long)gridDim.x * blockDim.x)
    p[i] = v;
}

__global__ void k_deg(const int* __restrict__ col, float* __restrict__ deg, int E) {
  for (long i = blockIdx.x * (long)blockDim.x + threadIdx.x; i < E;
       i += (long)gridDim.x * blockDim.x)
    atomicAdd(&deg[col[i]], 1.0f);
}

__global__ void k_rsqrt(float* __restrict__ d, int n) {
  int i = blockIdx.x * blockDim.x + threadIdx.x;
  if (i < n) d[i] = rsqrtf(d[i]);   // deg >= 1 always (self loop)
}

// f32 [rows, cin] -> f16 [rows, cpad], zero-padded columns
__global__ void k_cvt_pad(const float* __restrict__ src, _Float16* __restrict__ dst,
                          int rows, int cin, int cpad) {
  long total = (long)rows * cpad;
  for (long i = blockIdx.x * (long)blockDim.x + threadIdx.x; i < total;
       i += (long)gridDim.x * blockDim.x) {
    int r = (int)(i / cpad), c = (int)(i % cpad);
    dst[i] = (c < cin) ? (_Float16)src[(long)r * cin + c] : (_Float16)0.0f;
  }
}

// W [K, Nout] f32 -> Wt [Npad, Kpad] f16 transposed, zero padded
__global__ void k_cvt_wt(const float* __restrict__ W, _Float16* __restrict__ Wt,
                         int K, int Nout, int Kpad, int Npad) {
  long total = (long)Npad * Kpad;
  for (long i = blockIdx.x * (long)blockDim.x + threadIdx.x; i < total;
       i += (long)gridDim.x * blockDim.x) {
    int n = (int)(i / Kpad), k = (int)(i % Kpad);
    Wt[i] = (n < Nout && k < K) ? (_Float16)W[(long)k * Nout + n] : (_Float16)0.0f;
  }
}

// AGG[i,f] = dinv[i]^2 * H[i,f]   (self-loop contribution; also zero-inits AGG)
__global__ void k_selfinit(const float* __restrict__ H, const float* __restrict__ dinv,
                           float* __restrict__ AGG, int n, int F) {
  long total = (long)n * F;
  for (long i = blockIdx.x * (long)blockDim.x + threadIdx.x; i < total;
       i += (long)gridDim.x * blockDim.x) {
    int node = (int)(i / F);
    float d = dinv[node];
    AGG[i] = d * d * H[i];
  }
}

// AGG[col[e], f] += dinv[row]*dinv[col] * H[row[e], f]
__global__ void k_scatter(const int* __restrict__ row, const int* __restrict__ col,
                          const float* __restrict__ dinv, const float* __restrict__ H,
                          float* __restrict__ AGG, int E, int F) {
  long total = (long)E * F;
  for (long i = blockIdx.x * (long)blockDim.x + threadIdx.x; i < total;
       i += (long)gridDim.x * blockDim.x) {
    int e = (int)(i / F), f = (int)(i % F);
    int r = row[e], c = col[e];
    float nrm = dinv[r] * dinv[c];
    atomicAdd(&AGG[(long)c * F + f], nrm * H[(long)r * F + f]);
  }
}

__global__ void k_bias_relu(float* __restrict__ A, const float* __restrict__ b,
                            int n, int F) {
  long total = (long)n * F;
  for (long i = blockIdx.x * (long)blockDim.x + threadIdx.x; i < total;
       i += (long)gridDim.x * blockDim.x) {
    float v = A[i] + b[(int)(i % F)];
    A[i] = fmaxf(v, 0.0f);
  }
}

__global__ void k_pool_sum(const float* __restrict__ H, const int* __restrict__ batch,
                           float* __restrict__ gsum, int n, int F) {
  long total = (long)n * F;
  for (long i = blockIdx.x * (long)blockDim.x + threadIdx.x; i < total;
       i += (long)gridDim.x * blockDim.x) {
    int node = (int)(i / F), f = (int)(i % F);
    atomicAdd(&gsum[(long)batch[node] * F + f], H[i]);
  }
}

__global__ void k_pool_cnt(const int* __restrict__ batch, float* __restrict__ gcnt, int n) {
  for (long i = blockIdx.x * (long)blockDim.x + threadIdx.x; i < n;
       i += (long)gridDim.x * blockDim.x)
    atomicAdd(&gcnt[batch[i]], 1.0f);
}

__global__ void k_pool_div(float* __restrict__ gsum, const float* __restrict__ gcnt,
                           int G, int F) {
  long total = (long)G * F;
  for (long i = blockIdx.x * (long)blockDim.x + threadIdx.x; i < total;
       i += (long)gridDim.x * blockDim.x)
    gsum[i] /= fmaxf(gcnt[(int)(i / F)], 1.0f);
}

// ---------------- WMMA GEMM ----------------
// out[M, Nout] f32 = Xh[M, Kpad] f16 x Wt[Npad, Kpad]^T f16  (+bias, relu)
// One wave per 16x16 output tile, 8 waves per block.
// A layout (16-bit 16x32): lane<16 holds row (lane&15), K {kb+0..7, kb+16..23};
//                          lane>=16 holds K {kb+8..15, kb+24..31}.
// B layout (32x16):        lane<16 -> col (lane&15), K {kb+0..7, kb+16..23}; lane>=16 -> K {kb+8..15, kb+24..31}.
// C layout (f32 16x16):    VGPR j, lane<16 -> row j; lane>=16 -> row j+8; col = lane&15.
__global__ void k_wmma_gemm(const _Float16* __restrict__ Xh, const _Float16* __restrict__ Wt,
                            const float* __restrict__ bias, float* __restrict__ out,
                            int M, int Kpad, int Nout, int relu) {
  int mt = (M + 15) >> 4;
  int nt = (Nout + 15) >> 4;
  int tile = blockIdx.x * (blockDim.x >> 5) + (threadIdx.x >> 5);
  if (tile >= mt * nt) return;
  int tm = tile % mt;           // adjacent waves share the B column tile (L2 reuse)
  int tn = tile / mt;

  int lane = threadIdx.x & 31;
  int hs   = lane >> 4;         // half-select (which K sub-range this lane holds)
  int l15  = lane & 15;

  int r = tm * 16 + l15;
  bool rowok = (r < M);         // clamp reads; masked at store time
  const _Float16* arow = Xh + (size_t)(rowok ? r : 0) * Kpad;
  const _Float16* bcol = Wt + (size_t)(tn * 16 + l15) * Kpad;   // always < Npad rows

  v8f acc = {};
  for (int kb = 0; kb < Kpad; kb += 32) {
    v8h a0 = *(const v8h*)(arow + kb + hs * 8);        // K kb+hs*8   .. +7
    v8h a1 = *(const v8h*)(arow + kb + 16 + hs * 8);   // K kb+16+hs*8.. +7
    v8h b0 = *(const v8h*)(bcol + kb + hs * 8);        // K kb+hs*8   .. +7
    v8h b1 = *(const v8h*)(bcol + kb + 16 + hs * 8);   // K kb+16+hs*8.. +7
    v16h a, b;
#pragma unroll
    for (int i = 0; i < 8; ++i) {
      a[i] = a0[i]; a[i + 8] = a1[i];
      b[i] = b0[i]; b[i + 8] = b1[i];
    }
    acc = __builtin_amdgcn_wmma_f32_16x16x32_f16(
        /*neg_a=*/false, a, /*neg_b=*/false, b,
        /*c_mod=*/(short)0, acc, /*reuse_a=*/false, /*reuse_b=*/false);
  }

  int ocol = tn * 16 + l15;
  if (ocol < Nout) {
    float bv = bias ? bias[ocol] : 0.0f;
#pragma unroll
    for (int j = 0; j < 8; ++j) {
      int orow = tm * 16 + j + hs * 8;
      if (orow < M) {
        float v = acc[j] + bv;
        if (relu) v = fmaxf(v, 0.0f);
        out[(size_t)orow * Nout + ocol] = v;
      }
    }
  }
}

// ---------------- host-side orchestration ----------------

static inline int cdiv(long a, long b) { return (int)((a + b - 1) / b); }

extern "C" void kernel_launch(void* const* d_in, const int* in_sizes, int n_in,
                              void* d_out, int out_size, void* d_ws, size_t ws_size,
                              hipStream_t stream) {
  const float* x     = (const float*)d_in[0];
  const int*   ei    = (const int*)d_in[1];
  const int*   batch = (const int*)d_in[2];
  const float* W1  = (const float*)d_in[3];  const float* b1  = (const float*)d_in[4];
  const float* W2  = (const float*)d_in[5];  const float* b2  = (const float*)d_in[6];
  const float* W3  = (const float*)d_in[7];  const float* b3  = (const float*)d_in[8];
  const float* Wf1 = (const float*)d_in[9];  const float* bf1 = (const float*)d_in[10];
  const float* Wf2 = (const float*)d_in[11]; const float* bf2 = (const float*)d_in[12];
  float* outp = (float*)d_out;

  const int N = NNODES;
  const int E = in_sizes[1] / 2;
  const int* erow = ei;
  const int* ecol = ei + E;

  // ---- workspace carve-out (256B aligned) ----
  char* ws = (char*)d_ws;
  size_t off = 0;
  auto alloc = [&](size_t bytes) -> void* {
    void* p = ws + off;
    off += (bytes + 255) & ~(size_t)255;
    return p;
  };
  float*    dinv = (float*)alloc((size_t)N * 4);
  _Float16* XH   = (_Float16*)alloc((size_t)N * 128 * 2);   // padded f16 activations
  float*    H    = (float*)alloc((size_t)N * 216 * 4);      // GEMM output
  float*    AGG  = (float*)alloc((size_t)N * 216 * 4);      // scatter target
  _Float16* W1t  = (_Float16*)alloc((size_t)64 * 64 * 2);
  _Float16* W2t  = (_Float16*)alloc((size_t)112 * 64 * 2);
  _Float16* W3t  = (_Float16*)alloc((size_t)224 * 128 * 2);
  _Float16* Wf1t = (_Float16*)alloc((size_t)1024 * 224 * 2);
  _Float16* Wf2t = (_Float16*)alloc((size_t)128 * 1024 * 2);
  float*    gsum = (float*)alloc((size_t)NGRAPH * 216 * 4);
  float*    gcnt = (float*)alloc((size_t)NGRAPH * 4);
  _Float16* PH   = (_Float16*)alloc((size_t)NGRAPH * 224 * 2);
  float*    FC1  = (float*)alloc((size_t)NGRAPH * 1024 * 4);
  _Float16* FC1H = (_Float16*)alloc((size_t)NGRAPH * 1024 * 2);
  (void)ws_size;

  const int TB = 256;
  const int GS = 4096;      // grid-stride blocks, elementwise
  const int GSC = 8192;     // grid-stride blocks, scatter

  // ---- degree / dinv ----
  k_fill<<<cdiv(N, TB), TB, 0, stream>>>(dinv, 1.0f, N);   // self-loop => deg starts at 1
  k_deg<<<GS, TB, 0, stream>>>(ecol, dinv, E);
  k_rsqrt<<<cdiv(N, TB), TB, 0, stream>>>(dinv, N);

  // ---- weights -> f16 transposed/padded ----
  k_cvt_wt<<<cdiv((long)64 * 64, TB), TB, 0, stream>>>(W1, W1t, 54, 54, 64, 64);
  k_cvt_wt<<<cdiv((long)112 * 64, TB), TB, 0, stream>>>(W2, W2t, 54, 108, 64, 112);
  k_cvt_wt<<<cdiv((long)224 * 128, TB), TB, 0, stream>>>(W3, W3t, 108, 216, 128, 224);
  k_cvt_wt<<<cdiv((long)1024 * 224, TB), TB, 0, stream>>>(Wf1, Wf1t, 216, 1024, 224, 1024);
  k_cvt_wt<<<cdiv((long)128 * 1024, TB), TB, 0, stream>>>(Wf2, Wf2t, 1024, 128, 1024, 128);

  auto gemm = [&](const _Float16* A, const _Float16* Wt, const float* bias, float* O,
                  int M, int Kpad, int Nout, int relu) {
    long tiles = (long)cdiv(M, 16) * cdiv(Nout, 16);
    k_wmma_gemm<<<cdiv(tiles, 8), 256, 0, stream>>>(A, Wt, bias, O, M, Kpad, Nout, relu);
  };
  auto gcn_agg = [&](const float* bias, int F) {
    k_selfinit<<<GS, TB, 0, stream>>>(H, dinv, AGG, N, F);
    k_scatter<<<GSC, TB, 0, stream>>>(erow, ecol, dinv, H, AGG, E, F);
    k_bias_relu<<<GS, TB, 0, stream>>>(AGG, bias, N, F);
  };

  // ---- GCN layer 1: 54 -> 54 ----
  k_cvt_pad<<<GS, TB, 0, stream>>>(x, XH, N, 54, 64);
  gemm(XH, W1t, nullptr, H, N, 64, 54, 0);
  gcn_agg(b1, 54);
  k_cvt_pad<<<GS, TB, 0, stream>>>(AGG, XH, N, 54, 64);

  // ---- GCN layer 2: 54 -> 108 ----
  gemm(XH, W2t, nullptr, H, N, 64, 108, 0);
  gcn_agg(b2, 108);
  k_cvt_pad<<<GS, TB, 0, stream>>>(AGG, XH, N, 108, 128);

  // ---- GCN layer 3: 108 -> 216 ----
  gemm(XH, W3t, nullptr, H, N, 128, 216, 0);
  gcn_agg(b3, 216);   // AGG now holds h3 = relu(agg + b3), [N, 216]

  // ---- global mean pool ----
  k_fill<<<cdiv((long)NGRAPH * 216, TB), TB, 0, stream>>>(gsum, 0.0f, (long)NGRAPH * 216);
  k_fill<<<1, TB, 0, stream>>>(gcnt, 0.0f, NGRAPH);
  k_pool_sum<<<GS, TB, 0, stream>>>(AGG, batch, gsum, N, 216);
  k_pool_cnt<<<GS, TB, 0, stream>>>(batch, gcnt, N);
  k_pool_div<<<cdiv((long)NGRAPH * 216, TB), TB, 0, stream>>>(gsum, gcnt, NGRAPH, 216);
  k_cvt_pad<<<cdiv((long)NGRAPH * 224, TB), TB, 0, stream>>>(gsum, PH, NGRAPH, 216, 224);

  // ---- MLP head ----
  gemm(PH, Wf1t, bf1, FC1, NGRAPH, 224, 1024, 1);
  k_cvt_pad<<<cdiv((long)NGRAPH * 1024, TB), TB, 0, stream>>>(FC1, FC1H, NGRAPH, 1024, 1024);
  gemm(FC1H, Wf2t, bf2, outp, NGRAPH, 1024, 128, 0);
}
